// FP_MSDeformAttn_44702019617528
// MI455X (gfx1250) — matrix operand order repaired and checked by reading, not compiled, not run on hardware
//
#include <hip/hip_runtime.h>

// ---------------------------------------------------------------------------
// MS-Deformable-Attention for MI455X (gfx1250, wave32, WMMA)
//   N=2, LQ=17821, C=256, HD=8, L=4, P=4, D=32, K=256
// Pipeline:
//   0) pack each weight matrix once into bf16 WMMA B-fragment layout
//   1) value = input_flatten @ Wv + bv          (WMMA bf16 GEMM, Nout=256)
//   2) off   = query @ Woff + boff              (WMMA bf16 GEMM, Nout=256)
//   3) att   = query @ Watt + batt              (WMMA bf16 GEMM, Nout=128)
//   4) prep: softmax(att over 16), loc = ref + off/shape   (elementwise)
//   5) gather: one wave32 per (n,lq,head); lane = channel (D=32);
//      per-wave scalar coefficients via readfirstlane -> SMEM/SALU
//   6) out   = attn @ Wout + bout               (WMMA bf16 GEMM, Nout=256)
// ---------------------------------------------------------------------------

typedef __attribute__((ext_vector_type(16))) __bf16 v16bf;
typedef __attribute__((ext_vector_type(8)))  __bf16 v8bf;
typedef __attribute__((ext_vector_type(8)))  float  v8f;

#define N_  2
#define LQ_ 17821
#define C_  256
#define HD_ 8
#define L_  4
#define P_  4
#define D_  32
#define K_  256

__device__ __forceinline__ __bf16 f2bf(float f) {
  union { float f; unsigned u; } v; v.f = f;
  unsigned r = (v.u + 0x7FFFu + ((v.u >> 16) & 1u)) >> 16;   // round-nearest-even
  union { unsigned short s; __bf16 b; } o; o.s = (unsigned short)r;
  return o.b;
}

// ---------------------------------------------------------------------------
// Pack W[256 x Nout] (fp32 row-major) into bf16 WMMA B-fragments:
//   Bp[((nt*8 + kk)*32 + lane)*16 + e]  with the ISA 7.12.2 16-bit layout:
//   e<8  -> K = kk*32 + (lane>>4)*8 + e
//   e>=8 -> K = kk*32 + 16 + (lane>>4)*8 + (e-8),  N = nt*16 + (lane&15)
// Runs once per weight matrix; weights are tiny and L2-resident.
// ---------------------------------------------------------------------------
__global__ __launch_bounds__(256) void pack_b_kernel(
    const float* __restrict__ W, __bf16* __restrict__ Bp, int Nout)
{
  const int i = blockIdx.x * blockDim.x + threadIdx.x;
  if (i >= Nout * K_) return;
  const int e    = i & 15;
  const int lane = (i >> 4) & 31;
  const int kk   = (i >> 9) & 7;
  const int nt   = i >> 12;
  const int half = lane >> 4;
  const int k    = kk * 32 + ((e < 8) ? (half * 8 + e) : (16 + half * 8 + (e - 8)));
  const int ncol = nt * 16 + (lane & 15);
  Bp[i] = f2bf(W[(size_t)k * Nout + ncol]);
}

// ---------------------------------------------------------------------------
// GEMM: Y[M x NOUT] = X[M x 256] * Bp + bias; X fp32, Bp pre-packed bf16.
// Block = 256 threads (8 wave32), 64-row A panel staged in LDS as bf16.
// Inner loop: 2x ds_load_b128 (A) + 2x global_load_b128 (B) + 1 WMMA.
// NOUT is a template constant so all strides fold into shifts / IOFFSETs.
// ---------------------------------------------------------------------------
template <int NOUT>
__global__ __launch_bounds__(256) void gemm_bias_bf16_kernel(
    const float* __restrict__ X, const __bf16* __restrict__ Bp,
    const float* __restrict__ bias, float* __restrict__ Y, int M)
{
  // 64 rows x 256 cols bf16, padded to 264 (528 B row stride = 33*16 B:
  // 16-byte aligned v8bf reads, conflict-free across the 64 LDS banks).
  __shared__ __bf16 As[64 * 264];

  const int tid   = threadIdx.x;
  const int wave  = tid >> 5;
  const int lane  = tid & 31;
  const int mBase = blockIdx.x * 64;

  // Stage + convert the A panel into LDS (coalesced 1 KB per row).
  #pragma unroll 4
  for (int r = 0; r < 64; ++r) {
    const int row = mBase + r;
    const float x = (row < M) ? X[(size_t)row * K_ + tid] : 0.0f;
    As[r * 264 + tid] = f2bf(x);
  }
  __syncthreads();

  constexpr int nTiles   = NOUT >> 4;
  constexpr int totTiles = 4 * nTiles;   // 4 M-tiles per panel
  const int mrow = lane & 15;
  const int half = lane >> 4;

  for (int t = wave; t < totTiles; t += 8) {
    const int mt = t / nTiles;
    const int nt = t - mt * nTiles;
    const int ncol = nt * 16 + mrow;

    const __bf16* ab = &As[(mt * 16 + mrow) * 264];
    const __bf16* bb = Bp + ((size_t)(nt * 8) * 32 + lane) * 16;

    v8f acc = {};
    #pragma unroll
    for (int kk = 0; kk < K_ / 32; ++kk) {
      // A fragment: two 16 B LDS reads (constant ds offsets after unroll).
      const v8bf alo = *(const v8bf*)(ab + kk * 32 + half * 8);
      const v8bf ahi = *(const v8bf*)(ab + kk * 32 + 16 + half * 8);
      const v16bf a = __builtin_shufflevector(alo, ahi,
          0,1,2,3,4,5,6,7,8,9,10,11,12,13,14,15);

      // B fragment: two 16 B global reads, 1 KB coalesced per wave,
      // constant IOFFSET (kk*1024 B) off a single per-tile base.
      const v8bf blo = *(const v8bf*)(bb + kk * 512);
      const v8bf bhi = *(const v8bf*)(bb + kk * 512 + 8);
      const v16bf b = __builtin_shufflevector(blo, bhi,
          0,1,2,3,4,5,6,7,8,9,10,11,12,13,14,15);

      acc = __builtin_amdgcn_wmma_f32_16x16x32_bf16(
                false, a, false, b, (short)0, acc, false, false);
    }

    // Epilogue: D layout — lane gives N, VGPR r gives M = half*8 + r.
    // One base address per tile; stores at constant offsets r*NOUT*4 B.
    const float bcol = bias[ncol];
    const int row0 = mBase + mt * 16 + half * 8;
    float* yp = Y + (size_t)row0 * NOUT + ncol;
    if (row0 + 7 < M) {
      #pragma unroll
      for (int r = 0; r < 8; ++r) yp[r * NOUT] = acc[r] + bcol;
    } else {
      #pragma unroll
      for (int r = 0; r < 8; ++r)
        if (row0 + r < M) yp[r * NOUT] = acc[r] + bcol;
    }
  }
}

// ---------------------------------------------------------------------------
// Prep: softmax over the 16 (L*P) attention logits per (n,lq,head) and
// convert raw offsets (in-place) into normalized sampling locations.
// ---------------------------------------------------------------------------
__global__ __launch_bounds__(256) void prep_kernel(
    const float* __restrict__ refp, const int* __restrict__ shapes,
    float* __restrict__ loc, float* __restrict__ att)
{
  const int idx = blockIdx.x * blockDim.x + threadIdx.x;   // (n*LQ+lq)*HD+h
  if (idx >= N_ * LQ_ * HD_) return;
  const int nq = idx / HD_;                                // n*LQ + lq

  // softmax over 16
  float* a = att + (size_t)idx * (L_ * P_);
  float v[16], mx = -1e30f;
  #pragma unroll
  for (int i = 0; i < 16; ++i) { v[i] = a[i]; mx = fmaxf(mx, v[i]); }
  float s = 0.f;
  #pragma unroll
  for (int i = 0; i < 16; ++i) { v[i] = __expf(v[i] - mx); s += v[i]; }
  const float inv = 1.0f / s;
  #pragma unroll
  for (int i = 0; i < 16; ++i) a[i] = v[i] * inv;

  // loc = refpoint + offset / (W,H)
  float* o = loc + (size_t)idx * (L_ * P_ * 2);
  const float* rp = refp + (size_t)nq * (L_ * 2);
  #pragma unroll
  for (int l = 0; l < L_; ++l) {
    const float Wl = (float)shapes[l * 2 + 1];
    const float Hl = (float)shapes[l * 2 + 0];
    const float rx = rp[l * 2 + 0], ry = rp[l * 2 + 1];
    #pragma unroll
    for (int p = 0; p < P_; ++p) {
      const float ox = o[(l * P_ + p) * 2 + 0];
      const float oy = o[(l * P_ + p) * 2 + 1];
      o[(l * P_ + p) * 2 + 0] = rx + ox / Wl;
      o[(l * P_ + p) * 2 + 1] = ry + oy / Hl;
    }
  }
}

// ---------------------------------------------------------------------------
// Gather core: one wave32 per (n, lq, head); lane = channel (D == 32).
// wid is forced into an SGPR via readfirstlane so all bilinear coefficients,
// corner offsets and predicates become scalar (SMEM + SALU + s_cbranch);
// only the coalesced 128 B value fetches remain vector loads.
// ---------------------------------------------------------------------------
__global__ __launch_bounds__(256) void sample_kernel(
    const float* __restrict__ value, const float* __restrict__ loc,
    const float* __restrict__ att, const int* __restrict__ shapes,
    const int* __restrict__ starts, float* __restrict__ out, int Lin)
{
  const int wid0 = (blockIdx.x * blockDim.x + threadIdx.x) >> 5;  // (n*LQ+lq)*HD+h
  if (wid0 >= N_ * LQ_ * HD_) return;
  const int wid  = __builtin_amdgcn_readfirstlane(wid0);          // wave-uniform
  const int lane = threadIdx.x & 31;
  const int h  = wid & (HD_ - 1);
  const int nq = wid / HD_;
  const int n  = nq / LQ_;

  const float* lo = loc + (size_t)wid * (L_ * P_ * 2);
  const float* aw = att + (size_t)wid * (L_ * P_);

  float acc = 0.f;
  #pragma unroll
  for (int l = 0; l < L_; ++l) {
    const int Hl = shapes[l * 2 + 0], Wl = shapes[l * 2 + 1];
    const int s  = starts[l];
    const float* v = value + ((size_t)n * Lin + s) * C_ + h * D_ + lane;
    #pragma unroll
    for (int p = 0; p < P_; ++p) {
      const float lx = lo[(l * P_ + p) * 2 + 0];
      const float ly = lo[(l * P_ + p) * 2 + 1];
      const float a  = aw[l * P_ + p];
      const float x = lx * (float)Wl - 0.5f;
      const float y = ly * (float)Hl - 0.5f;
      const float xf = floorf(x), yf = floorf(y);
      const float fx = x - xf, fy = y - yf;
      const int x0 = (int)xf, y0 = (int)yf;
      const int x1 = x0 + 1,  y1 = y0 + 1;
      const float w00 = (1.f - fx) * (1.f - fy) * a;
      const float w10 = fx * (1.f - fy) * a;
      const float w01 = (1.f - fx) * fy * a;
      const float w11 = fx * fy * a;
      const bool vx0 = (x0 >= 0) & (x0 < Wl), vx1 = (x1 >= 0) & (x1 < Wl);
      const bool vy0 = (y0 >= 0) & (y0 < Hl), vy1 = (y1 >= 0) & (y1 < Hl);
      if (vx0 & vy0) acc += w00 * v[(size_t)(y0 * Wl + x0) * C_];
      if (vx1 & vy0) acc += w10 * v[(size_t)(y0 * Wl + x1) * C_];
      if (vx0 & vy1) acc += w01 * v[(size_t)(y1 * Wl + x0) * C_];
      if (vx1 & vy1) acc += w11 * v[(size_t)(y1 * Wl + x1) * C_];
    }
  }
  out[(size_t)nq * C_ + h * D_ + lane] = acc;
}

// ---------------------------------------------------------------------------
// Host-side launch
// ---------------------------------------------------------------------------
extern "C" void kernel_launch(void* const* d_in, const int* in_sizes, int n_in,
                              void* d_out, int out_size, void* d_ws, size_t ws_size,
                              hipStream_t stream) {
  const float* query  = (const float*)d_in[0];
  const float* refp   = (const float*)d_in[1];
  const float* inpf   = (const float*)d_in[2];
  const int*   shapes = (const int*)d_in[3];
  const int*   starts = (const int*)d_in[4];
  const float* Wv     = (const float*)d_in[5];
  const float* bv     = (const float*)d_in[6];
  const float* Woff   = (const float*)d_in[7];
  const float* boff   = (const float*)d_in[8];
  const float* Watt   = (const float*)d_in[9];
  const float* batt   = (const float*)d_in[10];
  const float* Wout   = (const float*)d_in[11];
  const float* bout   = (const float*)d_in[12];
  float* out = (float*)d_out;

  const int Lin = in_sizes[2] / (N_ * C_);
  const int Mv  = N_ * Lin;   // value GEMM rows
  const int Mq  = N_ * LQ_;   // query GEMM rows
  const int nqh = N_ * LQ_ * HD_;

  // Workspace layout (floats): value | loc(=off in-place) | att | attn_out,
  // then bf16 packed weights (all offsets are multiples of 1 KB -> aligned).
  float* ws    = (float*)d_ws;
  float* value = ws;
  float* locb  = value + (size_t)Mv * C_;
  float* attb  = locb  + (size_t)Mq * (HD_ * L_ * P_ * 2);
  float* ao    = attb  + (size_t)Mq * (HD_ * L_ * P_);
  __bf16* WvP   = (__bf16*)(ao + (size_t)Mq * C_);
  __bf16* WoffP = WvP   + 256 * 256;
  __bf16* WattP = WoffP + 256 * 256;
  __bf16* WoutP = WattP + 256 * 128;

  const dim3 blk(256);

  // 0) pack weights (once per call; trivial cost, L2-resident)
  pack_b_kernel<<<dim3((256 * 256 + 255) / 256), blk, 0, stream>>>(Wv,   WvP,   256);
  pack_b_kernel<<<dim3((256 * 256 + 255) / 256), blk, 0, stream>>>(Woff, WoffP, 256);
  pack_b_kernel<<<dim3((256 * 128 + 255) / 256), blk, 0, stream>>>(Watt, WattP, 128);
  pack_b_kernel<<<dim3((256 * 256 + 255) / 256), blk, 0, stream>>>(Wout, WoutP, 256);

  gemm_bias_bf16_kernel<256><<<dim3((Mv + 63) / 64), blk, 0, stream>>>(
      inpf, WvP, bv, value, Mv);
  gemm_bias_bf16_kernel<256><<<dim3((Mq + 63) / 64), blk, 0, stream>>>(
      query, WoffP, boff, locb, Mq);
  gemm_bias_bf16_kernel<128><<<dim3((Mq + 63) / 64), blk, 0, stream>>>(
      query, WattP, batt, attb, Mq);
  prep_kernel<<<dim3((nqh + 255) / 256), blk, 0, stream>>>(
      refp, shapes, locb, attb);
  sample_kernel<<<dim3((nqh * 32 + 255) / 256), blk, 0, stream>>>(
      value, locb, attb, shapes, starts, ao, Lin);
  gemm_bias_bf16_kernel<256><<<dim3((Mq + 63) / 64), blk, 0, stream>>>(
      ao, WoutP, bout, out, Mq);
}
